// MessagePassingLayer_66872640798743
// MI455X (gfx1250) — matrix operand order repaired and checked by reading, not compiled, not run on hardware
//
#include <hip/hip_runtime.h>

// MI455X / gfx1250, wave32. bf16 WMMA (v_wmma_f32_16x16x32_bf16) pipeline.
// B=8, N=2048, H=128.

#define GB 8
#define GN 2048
#define GH 128

typedef __attribute__((ext_vector_type(16))) __bf16 bf16x16;
typedef __attribute__((ext_vector_type(8)))  float  f32x8;

union FragAB { bf16x16 v; unsigned int u[8]; };
union FragC  { f32x8 v; float f[8]; };

// Packed f32->bf16 convert (RNE) — 1 VALU op per 2 elements.
__device__ __forceinline__ unsigned int pack2bf(float lo, float hi) {
  unsigned int r;
  asm("v_cvt_pk_bf16_f32 %0, %1, %2" : "=v"(r) : "v"(lo), "v"(hi));
  return r;
}
__device__ __forceinline__ unsigned short f2bf(float f) {
  return (unsigned short)(pack2bf(f, f) & 0xffffu);
}

// A-matrix (16x32 bf16) per-VGPR K base (ISA 7.12.2):
// VGPR0..3 -> K=2v (+8 hi half), VGPR4..7 -> K=16+2(v-4) (+8 hi half).
__device__ __forceinline__ int a_k0(int v, int half) {
  return 2 * v + 8 * half + ((v >= 4) ? 8 : 0);
}

__device__ __forceinline__ f32x8 wmma_bf16(const FragAB& a, const FragAB& b, f32x8 c) {
  return __builtin_amdgcn_wmma_f32_16x16x32_bf16(
      false, a.v, false, b.v, (short)0, c, false, false);
}

// 16x16 tile GEMM, K=128: A rows from LDS (bf16 pairs, given row stride in uints),
// B from pre-converted bf16 weight row (uint = packed K-pair).
__device__ __forceinline__ f32x8 gemm_lds_wb(const unsigned int* inLds,
                                             const unsigned int* Wrow,
                                             int half, int n, f32x8 acc) {
#pragma unroll
  for (int kk = 0; kk < 4; ++kk) {
    const int kbase = kk * 32;
    FragAB a, bm;
#pragma unroll
    for (int v = 0; v < 8; ++v) {
      a.u[v]  = inLds[n * 64 + ((kbase + a_k0(v, half)) >> 1)];
      bm.u[v] = Wrow[(kbase + a_k0(v, half)) >> 1];   // B 32x16: K=a_k0(v,half)
    }
    acc = wmma_bf16(a, bm, acc);
  }
  return acc;
}

// ---------------- Stage 0: convert W1,W2,W_ih,W_hh (f32) -> packed bf16 pairs
__global__ __launch_bounds__(256) void wconv_kernel(
    const float* __restrict__ W1, const float* __restrict__ W2,
    const float* __restrict__ Wih, const float* __restrict__ Whh,
    unsigned int* __restrict__ dst) {
  const int p = blockIdx.x * 256 + threadIdx.x;   // pair index, 65536 total
  const int e = p * 2;
  const float* src; int off;
  if      (e < 16384) { src = W1;  off = e; }
  else if (e < 32768) { src = W2;  off = e - 16384; }
  else if (e < 81920) { src = Wih; off = e - 32768; }
  else                { src = Whh; off = e - 81920; }
  dst[p] = pack2bf(src[off], src[off + 1]);
}

// ---------------- Stage 1: fused 2-layer MLP (Linear+ReLU x2) -> m2T bf16 (B,H,N)
__global__ __launch_bounds__(256) void mlp_kernel(
    const float* __restrict__ h,
    const unsigned int* __restrict__ W1b, const unsigned int* __restrict__ W2b,
    const float* __restrict__ b1, const float* __restrict__ b2,
    unsigned int* __restrict__ m2T_u) {
  __shared__ unsigned int hLds[16 * 64];    // 16 rows x 128 bf16 (packed pairs)
  __shared__ unsigned int m1Lds[16 * 64];

  const int b    = blockIdx.x >> 7;
  const int row0 = (blockIdx.x & 127) << 4;
  const int tid  = threadIdx.x;
  const int wave = tid >> 5;
  const int lane = tid & 31;
  const int half = lane >> 4;
  const int n    = lane & 15;
  const int col  = 16 * wave + n;

  // stage h tile: 512 float4 quads, 2 per thread
  const float4* hq = (const float4*)(h + ((size_t)b * GN + row0) * GH);
  for (int i = tid; i < 512; i += 256) {
    const float4 q = hq[i];
    hLds[2 * i]     = pack2bf(q.x, q.y);
    hLds[2 * i + 1] = pack2bf(q.z, q.w);
  }
  __syncthreads();

  const f32x8 z = {};
  // layer 1
  FragC acc1; acc1.v = gemm_lds_wb(hLds, W1b + (size_t)col * 64, half, n, z);
  {
    const float bias = b1[col];
    unsigned short* m1s = (unsigned short*)m1Lds;
#pragma unroll
    for (int i = 0; i < 8; ++i) {
      float val = acc1.f[i] + bias;
      val = val > 0.0f ? val : 0.0f;
      m1s[(i + 8 * half) * GH + col] = f2bf(val);
    }
  }
  __syncthreads();

  // layer 2 -> transposed store m2T[b][col][row0 + 8*half + 0..7], one b128
  FragC acc2; acc2.v = gemm_lds_wb(m1Lds, W2b + (size_t)col * 64, half, n, z);
  {
    const float bias = b2[col];
    float v[8];
#pragma unroll
    for (int i = 0; i < 8; ++i) {
      float t = acc2.f[i] + bias;
      v[i] = t > 0.0f ? t : 0.0f;
    }
    uint4 st;
    st.x = pack2bf(v[0], v[1]);
    st.y = pack2bf(v[2], v[3]);
    st.z = pack2bf(v[4], v[5]);
    st.w = pack2bf(v[6], v[7]);
    const size_t eidx = ((size_t)b * GH + col) * GN + row0 + 8 * half; // mult of 8
    *(uint4*)(m2T_u + (eidx >> 1)) = st;
  }
}

// ---------------- Stage 2: msg = relu(A @ m2), then GRU cell -> out f32
// Block = 32 rows x 128 cols. 8 waves; wave w owns cols 16w..16w+15,
// both 16-row sub-tiles. Double-buffered 32x64 bf16 adjacency chunks.
__global__ __launch_bounds__(256) void agg_gru_kernel(
    const float* __restrict__ h, const float* __restrict__ A,
    const unsigned int* __restrict__ m2u,
    const unsigned int* __restrict__ Wihb, const unsigned int* __restrict__ Whhb,
    const float* __restrict__ b_ih, const float* __restrict__ b_hh,
    float* __restrict__ out) {
  __shared__ unsigned int aLds[2][32 * 32];  // 32 rows x 64 bf16 cols, x2 buffers
  __shared__ unsigned int hLds[32 * 64];     // 32 x 128 bf16 h tile
  __shared__ unsigned int msgLds[32 * 64];   // 32 x 128 bf16 msg tile

  const int b    = blockIdx.x >> 6;          // 64 row-blocks per batch
  const int row0 = (blockIdx.x & 63) << 5;   // 32 rows per block
  const int tid  = threadIdx.x;
  const int wave = tid >> 5;
  const int lane = tid & 31;
  const int half = lane >> 4;
  const int n    = lane & 15;
  const int col  = 16 * wave + n;

  // stage h tile: 1024 float4 quads, 4 per thread
  const float4* hq = (const float4*)(h + ((size_t)b * GN + row0) * GH);
  for (int i = tid; i < 1024; i += 256) {
    const float4 q = hq[i];
    hLds[2 * i]     = pack2bf(q.x, q.y);
    hLds[2 * i + 1] = pack2bf(q.z, q.w);
  }

  // ---- aggregation: acc[rt] = A[32 x 2048] @ m2[2048 x 16 cols], 32 K-chunks of 64
  f32x8 acc0 = {}, acc1 = {};
  // cooperative A staging: thread handles float4 quads q = tid, tid+256
  // quad q -> row = q>>4, col quad = q&15
  const float4* Aq = (const float4*)(A + ((size_t)b * GN + row0) * GN);
  const size_t m2base = ((size_t)b * GH + col) * GN;   // bf16-element index

  {  // prologue: stage chunk 0  (chunk kk covers cols kk*64 .. kk*64+63)
#pragma unroll
    for (int j = 0; j < 2; ++j) {
      const int q = tid + 256 * j;
      const int r = q >> 4, cq = q & 15;
      const float4 f = Aq[(size_t)r * (GN / 4) + cq];
      aLds[0][r * 32 + 2 * cq]     = pack2bf(f.x, f.y);
      aLds[0][r * 32 + 2 * cq + 1] = pack2bf(f.z, f.w);
    }
  }
  __syncthreads();

#pragma unroll 2
  for (int kk = 0; kk < 32; ++kk) {
    const int cur = kk & 1;
    if (kk < 31) {                     // stage next 32x64 chunk into other buffer
#pragma unroll
      for (int j = 0; j < 2; ++j) {
        const int q = tid + 256 * j;
        const int r = q >> 4, cq = q & 15;
        const float4 f = Aq[(size_t)r * (GN / 4) + (kk + 1) * 16 + cq];
        aLds[cur ^ 1][r * 32 + 2 * cq]     = pack2bf(f.x, f.y);
        aLds[cur ^ 1][r * 32 + 2 * cq + 1] = pack2bf(f.z, f.w);
      }
    }
#pragma unroll
    for (int s = 0; s < 2; ++s) {      // two K=32 sub-steps per chunk
      FragAB bm;
#pragma unroll
      for (int v = 0; v < 8; ++v)
        bm.u[v] = m2u[(m2base + kk * 64 + 32 * s + a_k0(v, half)) >> 1];
      FragAB a0, a1;
#pragma unroll
      for (int v = 0; v < 8; ++v) {
        const int ko = 16 * s + (a_k0(v, half) >> 1);
        a0.u[v] = aLds[cur][n * 32 + ko];          // row-tile 0: rows 0..15
        a1.u[v] = aLds[cur][(16 + n) * 32 + ko];   // row-tile 1: rows 16..31
      }
      acc0 = wmma_bf16(a0, bm, acc0);
      acc1 = wmma_bf16(a1, bm, acc1);
    }
    __syncthreads();                   // chunk reads done; next buffer ready
  }

  // relu -> msg tile in LDS (bf16), both row-tiles
  {
    unsigned short* msgs = (unsigned short*)msgLds;
    FragC m0; m0.v = acc0;
    FragC m1; m1.v = acc1;
#pragma unroll
    for (int i = 0; i < 8; ++i) {
      const float v0 = m0.f[i] > 0.0f ? m0.f[i] : 0.0f;
      const float v1 = m1.f[i] > 0.0f ? m1.f[i] : 0.0f;
      msgs[(i + 8 * half) * GH + col]        = f2bf(v0);
      msgs[(16 + i + 8 * half) * GH + col]   = f2bf(v1);
    }
  }
  __syncthreads();

  // ---- GRU gates (torch order r,z,n over 3H rows of W_ih/W_hh), per row-tile
  const f32x8 z0 = {};
#pragma unroll
  for (int rt = 0; rt < 2; ++rt) {
    const unsigned int* msgT = msgLds + rt * 16 * 64;
    const unsigned int* hT   = hLds   + rt * 16 * 64;
    FragC sr, sz, xn, hn;
    sr.v = gemm_lds_wb(msgT, Wihb + (size_t)(0   + col) * 64, half, n, z0);
    sr.v = gemm_lds_wb(hT,   Whhb + (size_t)(0   + col) * 64, half, n, sr.v);
    sz.v = gemm_lds_wb(msgT, Wihb + (size_t)(128 + col) * 64, half, n, z0);
    sz.v = gemm_lds_wb(hT,   Whhb + (size_t)(128 + col) * 64, half, n, sz.v);
    xn.v = gemm_lds_wb(msgT, Wihb + (size_t)(256 + col) * 64, half, n, z0);
    hn.v = gemm_lds_wb(hT,   Whhb + (size_t)(256 + col) * 64, half, n, z0);

    const float br  = b_ih[col]       + b_hh[col];
    const float bz  = b_ih[128 + col] + b_hh[128 + col];
    const float bxn = b_ih[256 + col];
    const float bhn = b_hh[256 + col];

#pragma unroll
    for (int i = 0; i < 8; ++i) {
      const int M = rt * 16 + i + 8 * half;
      const size_t idx = ((size_t)b * GN + row0 + M) * GH + col;
      const float hv = h[idx];
      const float r  = 1.0f / (1.0f + __expf(-(sr.f[i] + br)));
      const float zz = 1.0f / (1.0f + __expf(-(sz.f[i] + bz)));
      const float nn = tanhf(xn.f[i] + bxn + r * (hn.f[i] + bhn));
      out[idx] = (1.0f - zz) * nn + zz * hv;
    }
  }
}

extern "C" void kernel_launch(void* const* d_in, const int* in_sizes, int n_in,
                              void* d_out, int out_size, void* d_ws, size_t ws_size,
                              hipStream_t stream) {
  const float* h    = (const float*)d_in[0];
  const float* A    = (const float*)d_in[1];
  const float* W1   = (const float*)d_in[2];
  const float* b1   = (const float*)d_in[3];
  const float* W2   = (const float*)d_in[4];
  const float* b2   = (const float*)d_in[5];
  const float* W_ih = (const float*)d_in[6];
  const float* W_hh = (const float*)d_in[7];
  const float* b_ih = (const float*)d_in[8];
  const float* b_hh = (const float*)d_in[9];

  // workspace: m2T bf16 (4 MB) then packed bf16 weights (256 KB)
  unsigned int* m2T_u = (unsigned int*)d_ws;
  unsigned int* wsW   = (unsigned int*)((char*)d_ws + (size_t)GB * GH * GN * 2);
  unsigned int* W1b   = wsW;             // 8192 uints
  unsigned int* W2b   = wsW + 8192;
  unsigned int* Wihb  = wsW + 16384;     // 24576 uints
  unsigned int* Whhb  = wsW + 40960;

  float* out = (float*)d_out;
  const dim3 block(256);

  wconv_kernel<<<dim3(256), block, 0, stream>>>(W1, W2, W_ih, W_hh, wsW);
  mlp_kernel<<<dim3(GB * (GN / 16)), block, 0, stream>>>(h, W1b, W2b, b1, b2, m2T_u);
  agg_gru_kernel<<<dim3(GB * (GN / 32)), block, 0, stream>>>(
      h, A, m2T_u, Wihb, Whhb, b_ih, b_hh, out);
}
